// SimpleSSMForecaster_34316788695596
// MI455X (gfx1250) — matrix operand order repaired and checked by reading, not compiled, not run on hardware
//
#include <hip/hip_runtime.h>
#include <hip/hip_bf16.h>

#define BATCHN 512
#define TLEN   512
#define INPD   128
#define HIDN   512
#define OUTD   3072          // INPD * HORIZON
#define SEGS   16
#define LSEG   (TLEN / SEGS) // 32
#define MT     32            // batch rows per workgroup
#define LDH    520           // padded LDS row stride (elements) -> conflict-friendly

typedef __attribute__((ext_vector_type(16))) __bf16 v16bf;
typedef __attribute__((ext_vector_type(8)))  __bf16 v8bf;
typedef __attribute__((ext_vector_type(8)))  float  v8f;

static __device__ __forceinline__ v8f wmma_bf16(v16bf a, v16bf b, v8f c) {
  // D = A(16x32) * B(32x16) + C, f32 accumulate
  return __builtin_amdgcn_wmma_f32_16x16x32_bf16(false, a, false, b, (short)0, c,
                                                 false, false);
}

// A-operand (16x32 bf16) from a row-major bf16 matrix, tile at (row0, k0).
// ISA layout: lanes 0-15 row M=lane, K {k0..k0+7, k0+16..k0+23};
//             lanes 16-31 row M=lane-16, K {k0+8..k0+15, k0+24..k0+31}.
static __device__ __forceinline__ v16bf load_a_bf16(const __bf16* base, int ld,
                                                    int row0, int k0, int lane) {
  const int half = lane >> 4;
  const int m    = lane & 15;
  const __bf16* p = base + (size_t)(row0 + m) * (size_t)ld + (size_t)k0 + 8 * half;
  v8bf lo = *(const v8bf*)(p);
  v8bf hi = *(const v8bf*)(p + 16);
  return __builtin_shufflevector(lo, hi, 0, 1, 2, 3, 4, 5, 6, 7,
                                         8, 9, 10, 11, 12, 13, 14, 15);
}

// Same A-operand but sourced from f32 memory (x), converting to bf16 on the fly.
static __device__ __forceinline__ v16bf load_a_f32(const float* base, size_t ld,
                                                   int row0, int k0, int lane) {
  const int half = lane >> 4;
  const int m    = lane & 15;
  const float* p = base + (size_t)(row0 + m) * ld + (size_t)k0 + 8 * half;
  v8f lo = *(const v8f*)(p);
  v8f hi = *(const v8f*)(p + 16);
  v16bf r;
#pragma unroll
  for (int i = 0; i < 8; ++i) { r[i] = (__bf16)lo[i]; r[i + 8] = (__bf16)hi[i]; }
  return r;
}

// B-operand (32x16 bf16) computing (W^T) tile: W row-major [n][k], tile (n0,k0).
// Lanes 0-15: column N=lane, K {k0..k0+7, k0+16..k0+23}; lanes 16-31: column N=lane-16, K {k0+8..k0+15, k0+24..k0+31}.
static __device__ __forceinline__ v16bf load_b_bf16(const __bf16* w, int ld,
                                                    int n0, int k0, int lane) {
  const int half = lane >> 4;
  const int n    = lane & 15;
  const __bf16* p = w + (size_t)(n0 + n) * (size_t)ld + (size_t)k0 + 8 * half;
  v8bf lo = *(const v8bf*)(p);
  v8bf hi = *(const v8bf*)(p + 16);
  return __builtin_shufflevector(lo, hi, 0, 1, 2, 3, 4, 5, 6, 7,
                                         8, 9, 10, 11, 12, 13, 14, 15);
}

__global__ void cvt_f32_bf16(const float* __restrict__ s, __bf16* __restrict__ d, int n) {
  int i = blockIdx.x * blockDim.x + threadIdx.x;
  if (i < n) d[i] = (__bf16)s[i];
}

// D = X @ X for 512x512 f32 (used for the A^L squaring chain; negligible cost)
__global__ void matsq_f32(const float* __restrict__ X, float* __restrict__ D) {
  int idx = blockIdx.x * blockDim.x + threadIdx.x;
  int m = idx >> 9, n = idx & 511;
  float s = 0.f;
  for (int k = 0; k < HIDN; ++k)
    s = fmaf(X[(size_t)m * HIDN + k], X[(size_t)k * HIDN + n], s);
  D[idx] = s;
}

// Phase 1: per (batch-tile, segment) compute c_s via local recurrence c <- c A^T + x_t B^T
__global__ __launch_bounds__(256) void ssm_phase1(const float* __restrict__ x,
                                                  const __bf16* __restrict__ Abf,
                                                  const __bf16* __restrict__ Bbf,
                                                  float* __restrict__ Cseg) {
  __shared__ __bf16 hbuf[MT * LDH];
  const int tid  = threadIdx.x;
  const int wave = tid >> 5;
  const int lane = tid & 31;
  const int b0   = blockIdx.x * MT;
  const int s    = blockIdx.y;
  const int n0w  = wave * 64;

  for (int i = tid; i < MT * LDH; i += 256) hbuf[i] = (__bf16)0.0f;
  __syncthreads();

  for (int j = 0; j < LSEG; ++j) {
    const int t = s * LSEG + j;
    v8f acc[2][4] = {};
    // c_{j-1} @ A^T   (keep rolled: avoids B-fragment register blowup/spills)
#pragma unroll 1
    for (int k0 = 0; k0 < HIDN; k0 += 32) {
      v16bf a0 = load_a_bf16(hbuf, LDH, 0, k0, lane);
      v16bf a1 = load_a_bf16(hbuf, LDH, 16, k0, lane);
#pragma unroll
      for (int nt = 0; nt < 4; ++nt) {
        v16bf bf = load_b_bf16(Abf, HIDN, n0w + nt * 16, k0, lane);
        acc[0][nt] = wmma_bf16(a0, bf, acc[0][nt]);
        acc[1][nt] = wmma_bf16(a1, bf, acc[1][nt]);
      }
    }
    // x_t @ B^T
    const float* xt = x + (size_t)b0 * TLEN * INPD + (size_t)t * INPD;
#pragma unroll 1
    for (int k0 = 0; k0 < INPD; k0 += 32) {
      v16bf a0 = load_a_f32(xt, (size_t)TLEN * INPD, 0, k0, lane);
      v16bf a1 = load_a_f32(xt, (size_t)TLEN * INPD, 16, k0, lane);
#pragma unroll
      for (int nt = 0; nt < 4; ++nt) {
        v16bf bf = load_b_bf16(Bbf, INPD, n0w + nt * 16, k0, lane);
        acc[0][nt] = wmma_bf16(a0, bf, acc[0][nt]);
        acc[1][nt] = wmma_bf16(a1, bf, acc[1][nt]);
      }
    }
    __syncthreads();  // all reads of hbuf done before overwrite
    const int n  = lane & 15;
    const int mb = (lane >> 4) * 8;
    if (j == LSEG - 1) {
      float* cs = Cseg + ((size_t)s * BATCHN + b0) * HIDN;
#pragma unroll
      for (int mt = 0; mt < 2; ++mt)
#pragma unroll
        for (int nt = 0; nt < 4; ++nt)
#pragma unroll
          for (int v = 0; v < 8; ++v)
            cs[(size_t)(mt * 16 + mb + v) * HIDN + n0w + nt * 16 + n] = acc[mt][nt][v];
    } else {
#pragma unroll
      for (int mt = 0; mt < 2; ++mt)
#pragma unroll
        for (int nt = 0; nt < 4; ++nt)
#pragma unroll
          for (int v = 0; v < 8; ++v)
            hbuf[(size_t)(mt * 16 + mb + v) * LDH + n0w + nt * 16 + n] =
                (__bf16)acc[mt][nt][v];
    }
    __syncthreads();
  }
}

// Phase 2: sequential combine h <- h (A^L)^T + c_s ; emits h_final as bf16
__global__ __launch_bounds__(256) void ssm_phase2(const __bf16* __restrict__ Pbf,
                                                  const float* __restrict__ Cseg,
                                                  __bf16* __restrict__ hfin) {
  __shared__ __bf16 hbuf[MT * LDH];
  const int tid  = threadIdx.x;
  const int wave = tid >> 5;
  const int lane = tid & 31;
  const int b0   = blockIdx.x * MT;
  const int n0w  = wave * 64;
  const int n    = lane & 15;
  const int mb   = (lane >> 4) * 8;

  for (int i = tid; i < MT * LDH; i += 256) hbuf[i] = (__bf16)0.0f;
  __syncthreads();

#pragma unroll 1
  for (int s = 0; s < SEGS; ++s) {
    v8f acc[2][4];
    const float* cs = Cseg + ((size_t)s * BATCHN + b0) * HIDN;
#pragma unroll
    for (int mt = 0; mt < 2; ++mt)
#pragma unroll
      for (int nt = 0; nt < 4; ++nt)
#pragma unroll
        for (int v = 0; v < 8; ++v)
          acc[mt][nt][v] = cs[(size_t)(mt * 16 + mb + v) * HIDN + n0w + nt * 16 + n];
#pragma unroll 1
    for (int k0 = 0; k0 < HIDN; k0 += 32) {
      v16bf a0 = load_a_bf16(hbuf, LDH, 0, k0, lane);
      v16bf a1 = load_a_bf16(hbuf, LDH, 16, k0, lane);
#pragma unroll
      for (int nt = 0; nt < 4; ++nt) {
        v16bf bf = load_b_bf16(Pbf, HIDN, n0w + nt * 16, k0, lane);
        acc[0][nt] = wmma_bf16(a0, bf, acc[0][nt]);
        acc[1][nt] = wmma_bf16(a1, bf, acc[1][nt]);
      }
    }
    __syncthreads();
    if (s == SEGS - 1) {
#pragma unroll
      for (int mt = 0; mt < 2; ++mt)
#pragma unroll
        for (int nt = 0; nt < 4; ++nt)
#pragma unroll
          for (int v = 0; v < 8; ++v)
            hfin[(size_t)(b0 + mt * 16 + mb + v) * HIDN + n0w + nt * 16 + n] =
                (__bf16)acc[mt][nt][v];
    } else {
#pragma unroll
      for (int mt = 0; mt < 2; ++mt)
#pragma unroll
        for (int nt = 0; nt < 4; ++nt)
#pragma unroll
          for (int v = 0; v < 8; ++v)
            hbuf[(size_t)(mt * 16 + mb + v) * LDH + n0w + nt * 16 + n] =
                (__bf16)acc[mt][nt][v];
    }
    __syncthreads();
  }
}

// out = h_final @ Wc^T + bc   (M=512, N=3072, K=512)
__global__ __launch_bounds__(256) void ssm_final(const __bf16* __restrict__ hbf,
                                                 const __bf16* __restrict__ Wcbf,
                                                 const float* __restrict__ bc,
                                                 float* __restrict__ out) {
  const int tid  = threadIdx.x;
  const int wave = tid >> 5;
  const int lane = tid & 31;
  const int b0   = blockIdx.x * MT;
  const int n0   = blockIdx.y * 512 + wave * 64;
  v8f acc[2][4] = {};
#pragma unroll 1
  for (int k0 = 0; k0 < HIDN; k0 += 32) {
    v16bf a0 = load_a_bf16(hbf, HIDN, b0, k0, lane);
    v16bf a1 = load_a_bf16(hbf, HIDN, b0 + 16, k0, lane);
#pragma unroll
    for (int nt = 0; nt < 4; ++nt) {
      v16bf bf = load_b_bf16(Wcbf, HIDN, n0 + nt * 16, k0, lane);
      acc[0][nt] = wmma_bf16(a0, bf, acc[0][nt]);
      acc[1][nt] = wmma_bf16(a1, bf, acc[1][nt]);
    }
  }
  const int n  = lane & 15;
  const int mb = (lane >> 4) * 8;
#pragma unroll
  for (int nt = 0; nt < 4; ++nt) {
    const int col  = n0 + nt * 16 + n;
    const float bv = bc[col];
#pragma unroll
    for (int mt = 0; mt < 2; ++mt)
#pragma unroll
      for (int v = 0; v < 8; ++v)
        out[(size_t)(b0 + mt * 16 + mb + v) * OUTD + col] = acc[mt][nt][v] + bv;
  }
}

extern "C" void kernel_launch(void* const* d_in, const int* in_sizes, int n_in,
                              void* d_out, int out_size, void* d_ws, size_t ws_size,
                              hipStream_t stream) {
  (void)in_sizes; (void)n_in; (void)out_size; (void)ws_size;
  const float* x  = (const float*)d_in[0];
  const float* A  = (const float*)d_in[1];
  const float* B  = (const float*)d_in[2];
  const float* Wc = (const float*)d_in[3];
  const float* bc = (const float*)d_in[4];
  float* out = (float*)d_out;

  char* ws = (char*)d_ws;
  size_t off = 0;
  auto take = [&](size_t bytes) {
    char* p = ws + off;
    off = (off + bytes + 255) & ~(size_t)255;
    return p;
  };
  __bf16* Abf  = (__bf16*)take((size_t)HIDN * HIDN * 2);
  __bf16* Bbf  = (__bf16*)take((size_t)HIDN * INPD * 2);
  __bf16* Wcbf = (__bf16*)take((size_t)OUTD * HIDN * 2);
  __bf16* Pbf  = (__bf16*)take((size_t)HIDN * HIDN * 2);
  float*  W1   = (float*)take((size_t)HIDN * HIDN * 4);
  float*  W2   = (float*)take((size_t)HIDN * HIDN * 4);
  float*  Cseg = (float*)take((size_t)SEGS * BATCHN * HIDN * 4);
  __bf16* hfin = (__bf16*)take((size_t)BATCHN * HIDN * 2);

  cvt_f32_bf16<<<(HIDN * HIDN + 255) / 256, 256, 0, stream>>>(A, Abf, HIDN * HIDN);
  cvt_f32_bf16<<<(HIDN * INPD + 255) / 256, 256, 0, stream>>>(B, Bbf, HIDN * INPD);
  cvt_f32_bf16<<<(OUTD * HIDN + 255) / 256, 256, 0, stream>>>(Wc, Wcbf, OUTD * HIDN);
  // A^32 = ((((A^2)^2)^2)^2)^2  (f32 chain, tiny cost)
  matsq_f32<<<HIDN * HIDN / 256, 256, 0, stream>>>(A,  W1);  // A^2
  matsq_f32<<<HIDN * HIDN / 256, 256, 0, stream>>>(W1, W2);  // A^4
  matsq_f32<<<HIDN * HIDN / 256, 256, 0, stream>>>(W2, W1);  // A^8
  matsq_f32<<<HIDN * HIDN / 256, 256, 0, stream>>>(W1, W2);  // A^16
  matsq_f32<<<HIDN * HIDN / 256, 256, 0, stream>>>(W2, W1);  // A^32 == A^LSEG
  cvt_f32_bf16<<<(HIDN * HIDN + 255) / 256, 256, 0, stream>>>(W1, Pbf, HIDN * HIDN);

  ssm_phase1<<<dim3(BATCHN / MT, SEGS), 256, 0, stream>>>(x, Abf, Bbf, Cseg);
  ssm_phase2<<<BATCHN / MT, 256, 0, stream>>>(Pbf, Cseg, hfin);
  ssm_final<<<dim3(BATCHN / MT, OUTD / 512), 256, 0, stream>>>(hfin, Wcbf, bc, out);
}
